// TransformerModel_16896401342687
// MI455X (gfx1250) — compile-verified, matrix operand order checked by reading, not run on hardware
//
#include <hip/hip_runtime.h>
#include <math.h>
#include <stdint.h>

// ---------------------------------------------------------------------------
// Types for CDNA5 WMMA (wave32): v_wmma_f32_16x16x32_bf16
// ---------------------------------------------------------------------------
typedef __attribute__((ext_vector_type(16))) __bf16 v16bf;
typedef __attribute__((ext_vector_type(8)))  float  v8f;

union ABFrag { uint4 u[2]; v16bf v; };

__device__ __forceinline__ unsigned short f2bf(float x) {
  unsigned u = __float_as_uint(x);
  u += 0x7FFFu + ((u >> 16) & 1u);          // round-to-nearest-even
  return (unsigned short)(u >> 16);
}

// ---------------------------------------------------------------------------
// f32 -> bf16 with K zero-padding:  src[rows][Ks] -> dst[rows][Kd]
// ---------------------------------------------------------------------------
__global__ void k_cvt_bf16(const float* __restrict__ src, unsigned short* __restrict__ dst,
                           int rows, int Ks, int Kd) {
  int i = blockIdx.x * blockDim.x + threadIdx.x;
  if (i >= rows * Kd) return;
  int r = i / Kd, k = i - r * Kd;
  dst[i] = (k < Ks) ? f2bf(src[(size_t)r * Ks + k]) : (unsigned short)0;
}

// ---------------------------------------------------------------------------
// Pack 4 weight matrices [K][Cout] into fused transposed bf16 Bt[Np][Kp]
// (N-major, K-contiguous) + fused bias[Np]; pads are zero.
// ---------------------------------------------------------------------------
__global__ void k_pack_w(const float* __restrict__ W0, const float* __restrict__ W1,
                         const float* __restrict__ W2, const float* __restrict__ W3,
                         const float* __restrict__ b0, const float* __restrict__ b1,
                         const float* __restrict__ b2, const float* __restrict__ b3,
                         unsigned short* __restrict__ Bt, float* __restrict__ biasOut,
                         int K, int Kp, int Cout, int Np) {
  int i = blockIdx.x * blockDim.x + threadIdx.x;
  if (i >= Np * Kp) return;
  int n = i / Kp, k = i - n * Kp;
  int mat = n / Cout, c = n - mat * Cout;
  const float* W = nullptr; const float* b = nullptr;
  if (mat == 0) { W = W0; b = b0; } else if (mat == 1) { W = W1; b = b1; }
  else if (mat == 2) { W = W2; b = b2; } else if (mat == 3) { W = W3; b = b3; }
  unsigned short v = 0;
  if (W && k < K) v = f2bf(W[(size_t)k * Cout + c]);
  Bt[i] = v;
  if (k == 0) biasOut[n] = W ? b[c] : 0.0f;
}

// ---------------------------------------------------------------------------
// WMMA GEMM:  C[M][ldc] = A[M][Kp](bf16) * Bt[ldc][Kp]^T(bf16) + bias[ldc]
// Block = 8 waves = 128 rows x 64 cols. The 64xKp B tile is shared by all
// waves -> staged in LDS once (padded stride for bank-conflict-free b128
// reads). Both the A fragment (global) and the B fragment (LDS) are
// software ping-ponged so each WMMA's operands were requested one step
// earlier and their latency is hidden under the previous WMMA.
// Kp multiple of 32; ldc multiple of 64; blockDim = 256.
// ---------------------------------------------------------------------------
__global__ void k_wmma_gemm(const unsigned short* __restrict__ A,
                            const unsigned short* __restrict__ Bt,
                            const float* __restrict__ bias,
                            float* __restrict__ C,
                            int M, int Kp, int ldc) {
  extern __shared__ unsigned short sB[];        // [64][Kp+8]
  const int ls = Kp + 8;                        // padded LDS row stride (shorts)
  const int n0 = blockIdx.y * 64;

  // Cooperative load of the 64 x Kp B tile into LDS (16B chunks).
  {
    const int cpk = Kp / 8;                     // 16B chunks per row
    const int chunks = 64 * cpk;
    for (int id = threadIdx.x; id < chunks; id += blockDim.x) {
      int r = id / cpk, c8 = id - r * cpk;
      *(uint4*)(sB + r * ls + c8 * 8) =
          *(const uint4*)(Bt + (size_t)(n0 + r) * Kp + c8 * 8);
    }
  }
  __syncthreads();

  const int lane  = threadIdx.x & 31;
  const int wave  = threadIdx.x >> 5;
  const int tm    = blockIdx.x * 8 + wave;      // 16-row tile index
  if (tm * 16 >= M) return;                     // uniform per wave: EXEC all-1s
  const int nlane = lane & 15;
  const int half  = lane >> 4;
  const int kb    = half * 8;

  int arowi = tm * 16 + nlane; if (arowi >= M) arowi = M - 1;  // clamp keeps EXEC full
  const unsigned short* arow = A + (size_t)arowi * Kp;

  const unsigned short* brow[4];
#pragma unroll
  for (int t = 0; t < 4; ++t)
    brow[t] = sB + (t * 16 + nlane) * ls;

  v8f acc[4] = {};

  // Preload first A fragment (global) and first B fragment (LDS).
  ABFrag a, bc;
  a.u[0]  = *(const uint4*)(arow + kb);
  a.u[1]  = *(const uint4*)(arow + 16 + kb);
  bc.u[0] = *(const uint4*)(brow[0] + kb);
  bc.u[1] = *(const uint4*)(brow[0] + 16 + kb);

  for (int k0 = 0; k0 < Kp; k0 += 32) {
    ABFrag an = a;                              // prefetch next A chunk (global)
    if (k0 + 32 < Kp) {
      an.u[0] = *(const uint4*)(arow + k0 + 32 + kb);
      an.u[1] = *(const uint4*)(arow + k0 + 48 + kb);
    }
#pragma unroll
    for (int t = 0; t < 4; ++t) {
      ABFrag bn = bc;                           // prefetch next B fragment (LDS)
      if (t < 3) {
        bn.u[0] = *(const uint4*)(brow[t + 1] + k0 + kb);
        bn.u[1] = *(const uint4*)(brow[t + 1] + k0 + 16 + kb);
      } else if (k0 + 32 < Kp) {
        bn.u[0] = *(const uint4*)(brow[0] + k0 + 32 + kb);
        bn.u[1] = *(const uint4*)(brow[0] + k0 + 48 + kb);
      }
      acc[t] = __builtin_amdgcn_wmma_f32_16x16x32_bf16(
          /*neg_a=*/false, a.v, /*neg_b=*/false, bc.v,
          /*c_mod=*/(short)0, acc[t], /*reuse_a=*/false, /*reuse_b=*/false);
      bc = bn;
    }
    a = an;
  }

#pragma unroll
  for (int t = 0; t < 4; ++t) {
    const int col = n0 + t * 16 + nlane;
    const float bv = bias[col];
#pragma unroll
    for (int i = 0; i < 8; ++i) {
      const int m = tm * 16 + 8 * half + i;
      if (m < M) C[(size_t)m * ldc + col] = acc[t][i] + bv;
    }
  }
}

// ---------------------------------------------------------------------------
// CSR build over dst (shared by both layers; integer atomics only)
// ---------------------------------------------------------------------------
__global__ void k_count(const int* __restrict__ dst, int* __restrict__ cnt, int E) {
  int i = blockIdx.x * blockDim.x + threadIdx.x;
  if (i < E) atomicAdd(&cnt[dst[i]], 1);
}

// single block, 256 threads: exclusive scan cnt[0..n) -> rowstart[0..n]
__global__ void k_scan(const int* __restrict__ cnt, int* __restrict__ rowstart, int n) {
  __shared__ int s[256];
  __shared__ int carry;
  int tid = threadIdx.x;
  if (tid == 0) { carry = 0; rowstart[0] = 0; }
  __syncthreads();
  for (int base = 0; base < n; base += 256) {
    int v = (base + tid < n) ? cnt[base + tid] : 0;
    s[tid] = v; __syncthreads();
    for (int off = 1; off < 256; off <<= 1) {
      int t = (tid >= off) ? s[tid - off] : 0;
      __syncthreads();
      s[tid] += t;
      __syncthreads();
    }
    if (base + tid < n) rowstart[base + tid + 1] = carry + s[tid];
    __syncthreads();
    if (tid == 0) carry += s[255];
    __syncthreads();
  }
}

__global__ void k_fill(const int* __restrict__ dst, int* __restrict__ cursor,
                       int* __restrict__ eidx, int E) {
  int i = blockIdx.x * blockDim.x + threadIdx.x;
  if (i >= E) return;
  int pos = atomicAdd(&cursor[dst[i]], 1);
  eidx[pos] = i;
}

// ---------------------------------------------------------------------------
// Edge attention. Fused feature buffer F[node][ldF]:
//   q: cols [0,2C)   k: [2C,4C)   v: [4C,6C)   skip: [6C,8C)   (heads=2)
// ---------------------------------------------------------------------------
__global__ void k_edge_logits(const float* __restrict__ F, int ldF, int C, float scale,
                              const int* __restrict__ src, const int* __restrict__ dst, int E,
                              float* __restrict__ alpha) {
  int gw   = (blockIdx.x * blockDim.x + threadIdx.x) >> 5;   // global wave id
  int lane = threadIdx.x & 31;
  int e = gw >> 1, h = gw & 1;
  if (e >= E) return;
  int s = src[e], d = dst[e];
  const float* q = F + (size_t)d * ldF + h * C;
  const float* k = F + (size_t)s * ldF + 2 * C + h * C;
  float sum = 0.0f;
  for (int c = lane; c < C; c += 32) sum += q[c] * k[c];
#pragma unroll
  for (int o = 16; o > 0; o >>= 1) sum += __shfl_xor(sum, o, 32);
  if (lane == 0) alpha[e * 2 + h] = sum * scale;
}

// per (node, head): softmax max + denominator over the node's incoming edges
__global__ void k_node_stats(const float* __restrict__ alpha, const int* __restrict__ rs,
                             const int* __restrict__ eidx,
                             float* __restrict__ nm, float* __restrict__ den, int Nn) {
  int i = blockIdx.x * blockDim.x + threadIdx.x;
  if (i >= Nn * 2) return;
  int d = i >> 1, h = i & 1;
  int j0 = rs[d], j1 = rs[d + 1];
  float m = -3.4e38f;
  for (int j = j0; j < j1; ++j) m = fmaxf(m, alpha[eidx[j] * 2 + h]);
  float s = 0.0f;
  for (int j = j0; j < j1; ++j) s += __expf(alpha[eidx[j] * 2 + h] - m);
  nm[i] = m;
  den[i] = (s > 0.0f) ? s : 1.0f;
}

__global__ void k_edge_prob(const float* __restrict__ alpha, const int* __restrict__ dst,
                            const float* __restrict__ nm, const float* __restrict__ den,
                            float* __restrict__ prob, int E) {
  int i = blockIdx.x * blockDim.x + threadIdx.x;
  if (i >= E * 2) return;
  int e = i >> 1, h = i & 1;
  int d = dst[e];
  prob[i] = __expf(alpha[i] - nm[d * 2 + h]) / den[d * 2 + h];
}

// per (node, head) block: gather-aggregate (no atomics) + fused skip add
// out[n][2C]: cols h*C + c
__global__ void k_node_agg(const float* __restrict__ F, int ldF, int C,
                           const int* __restrict__ rs, const int* __restrict__ eidx,
                           const float* __restrict__ prob, const int* __restrict__ src,
                           float* __restrict__ out) {
  int d = blockIdx.x, h = blockIdx.y;
  int tid = threadIdx.x;
  float acc[5] = {0.f, 0.f, 0.f, 0.f, 0.f};
  int j0 = rs[d], j1 = rs[d + 1];
  for (int j = j0; j < j1; ++j) {
    int e = eidx[j];
    float p = prob[e * 2 + h];
    const float* vr = F + (size_t)src[e] * ldF + 4 * C + h * C;
#pragma unroll
    for (int r = 0; r < 5; ++r) {
      int c = tid + r * blockDim.x;
      if (c < C) acc[r] += p * vr[c];
    }
  }
  const float* sk = F + (size_t)d * ldF + 6 * C + h * C;
  float* o = out + (size_t)d * (2 * C) + h * C;
#pragma unroll
  for (int r = 0; r < 5; ++r) {
    int c = tid + r * blockDim.x;
    if (c < C) o[c] = acc[r] + sk[c];
  }
}

// out[n][c] = max(h2[n][2c], h2[n][2c+1])  (reshape(N,550,2) max over last axis)
__global__ void k_final(const float* __restrict__ h2, float* __restrict__ out, int Nn) {
  int i = blockIdx.x * blockDim.x + threadIdx.x;
  if (i >= Nn * 550) return;
  int n = i / 550, c = i - n * 550;
  const float* r = h2 + (size_t)n * 1100 + 2 * c;
  out[i] = fmaxf(r[0], r[1]);
}

// ---------------------------------------------------------------------------
// Launch
// ---------------------------------------------------------------------------
extern "C" void kernel_launch(void* const* d_in, const int* in_sizes, int n_in,
                              void* d_out, int out_size, void* d_ws, size_t ws_size,
                              hipStream_t stream) {
  const float* x  = (const float*)d_in[0];
  const int*   ei = (const int*)d_in[1];
  const int Nn = in_sizes[0] / 256;     // 20000
  const int E  = in_sizes[1] / 2;       // 120000
  const int* src = ei;
  const int* dst = ei + E;

  const float *Wq1=(const float*)d_in[2],  *bq1=(const float*)d_in[3];
  const float *Wk1=(const float*)d_in[4],  *bk1=(const float*)d_in[5];
  const float *Wv1=(const float*)d_in[6],  *bv1=(const float*)d_in[7];
  const float *Ws1=(const float*)d_in[8],  *bs1=(const float*)d_in[9];
  const float *Wq2=(const float*)d_in[10], *bq2=(const float*)d_in[11];
  const float *Wk2=(const float*)d_in[12], *bk2=(const float*)d_in[13];
  const float *Wv2=(const float*)d_in[14], *bv2=(const float*)d_in[15];
  const float *Ws2=(const float*)d_in[16], *bs2=(const float*)d_in[17];

  const int K1 = 256, Kp1 = 256, Np1 = 320;   // fused 4*66  = 264 -> 320
  const int K2 = 66,  Kp2 = 96,  Np2 = 4416;  // fused 4*1100 = 4400 -> 4416

  // Workspace carve (256B aligned)
  char* p = (char*)d_ws;
  auto alloc = [&](size_t b) -> void* { void* r = p; p += (b + 255) & ~(size_t)255; return r; };
  unsigned short* xb    = (unsigned short*)alloc((size_t)Nn * Kp1 * 2);
  unsigned short* Bt1   = (unsigned short*)alloc((size_t)Np1 * Kp1 * 2);
  float*          bias1 = (float*)alloc((size_t)Np1 * 4);
  float*          qkvs1 = (float*)alloc((size_t)Nn * Np1 * 4);
  float*          h1f   = (float*)alloc((size_t)Nn * 66 * 4);
  unsigned short* h1b   = (unsigned short*)alloc((size_t)Nn * Kp2 * 2);
  unsigned short* Bt2   = (unsigned short*)alloc((size_t)Np2 * Kp2 * 2);
  float*          bias2 = (float*)alloc((size_t)Np2 * 4);
  float*          qkvs2 = (float*)alloc((size_t)Nn * Np2 * 4);
  float*          h2    = (float*)alloc((size_t)Nn * 1100 * 4);
  float*          alpha = (float*)alloc((size_t)E * 2 * 4);
  float*          prob  = (float*)alloc((size_t)E * 2 * 4);
  float*          nm    = (float*)alloc((size_t)Nn * 2 * 4);
  float*          den   = (float*)alloc((size_t)Nn * 2 * 4);
  int*            cnt   = (int*)alloc((size_t)Nn * 4);
  int*            rowst = (int*)alloc((size_t)(Nn + 1) * 4);
  int*            cursor= (int*)alloc((size_t)Nn * 4);
  int*            eidx  = (int*)alloc((size_t)E * 4);

  // ---- CSR over dst (shared by both layers; only int atomics) ----
  hipMemsetAsync(cnt, 0, (size_t)Nn * 4, stream);
  k_count<<<(E + 255) / 256, 256, 0, stream>>>(dst, cnt, E);
  k_scan<<<1, 256, 0, stream>>>(cnt, rowst, Nn);
  hipMemcpyAsync(cursor, rowst, (size_t)Nn * 4, hipMemcpyDeviceToDevice, stream);
  k_fill<<<(E + 255) / 256, 256, 0, stream>>>(dst, cursor, eidx, E);

  // ---- Layer 1 ----
  k_cvt_bf16<<<(Nn * Kp1 + 255) / 256, 256, 0, stream>>>(x, xb, Nn, K1, Kp1);
  k_pack_w<<<(Np1 * Kp1 + 255) / 256, 256, 0, stream>>>(
      Wq1, Wk1, Wv1, Ws1, bq1, bk1, bv1, bs1, Bt1, bias1, K1, Kp1, 66, Np1);
  {
    dim3 g((Nn / 16 + 7) / 8, Np1 / 64);
    size_t shm = (size_t)64 * (Kp1 + 8) * 2;
    k_wmma_gemm<<<g, 256, shm, stream>>>(xb, Bt1, bias1, qkvs1, Nn, Kp1, Np1);
  }
  k_edge_logits<<<(E * 2 * 32 + 255) / 256, 256, 0, stream>>>(
      qkvs1, Np1, 33, 1.0f / sqrtf(33.0f), src, dst, E, alpha);
  k_node_stats<<<(Nn * 2 + 255) / 256, 256, 0, stream>>>(alpha, rowst, eidx, nm, den, Nn);
  k_edge_prob<<<(E * 2 + 255) / 256, 256, 0, stream>>>(alpha, dst, nm, den, prob, E);
  k_node_agg<<<dim3(Nn, 2), 64, 0, stream>>>(qkvs1, Np1, 33, rowst, eidx, prob, src, h1f);
  k_cvt_bf16<<<(Nn * Kp2 + 255) / 256, 256, 0, stream>>>(h1f, h1b, Nn, 66, Kp2);

  // ---- Layer 2 ----
  k_pack_w<<<(Np2 * Kp2 + 255) / 256, 256, 0, stream>>>(
      Wq2, Wk2, Wv2, Ws2, bq2, bk2, bv2, bs2, Bt2, bias2, K2, Kp2, 1100, Np2);
  {
    dim3 g((Nn / 16 + 7) / 8, Np2 / 64);
    size_t shm = (size_t)64 * (Kp2 + 8) * 2;
    k_wmma_gemm<<<g, 256, shm, stream>>>(h1b, Bt2, bias2, qkvs2, Nn, Kp2, Np2);
  }
  k_edge_logits<<<(E * 2 * 32 + 255) / 256, 256, 0, stream>>>(
      qkvs2, Np2, 550, 1.0f / sqrtf(550.0f), src, dst, E, alpha);
  k_node_stats<<<(Nn * 2 + 255) / 256, 256, 0, stream>>>(alpha, rowst, eidx, nm, den, Nn);
  k_edge_prob<<<(E * 2 + 255) / 256, 256, 0, stream>>>(alpha, dst, nm, den, prob, E);
  k_node_agg<<<dim3(Nn, 2), 128, 0, stream>>>(qkvs2, Np2, 550, rowst, eidx, prob, src, h2);

  k_final<<<(Nn * 550 + 255) / 256, 256, 0, stream>>>(h2, (float*)d_out, Nn);
}
